// LstmLayer_80951543595849
// MI455X (gfx1250) — matrix operand order repaired and checked by reading, not compiled, not run on hardware
//
#include <hip/hip_runtime.h>

// ---------------------------------------------------------------------------
// LSTM (B=64, T=512, D=512, U=1024) for gfx1250 / MI455X.
// bf16 WMMA (V_WMMA_F32_16X16X32_BF16), weights pinned in 192KB LDS per WGP,
// persistent kernel, grid barrier per timestep, cell state c in VGPRs,
// hardware v_tanh_f32 activations.
// ---------------------------------------------------------------------------

typedef __bf16 bf16;
typedef __attribute__((ext_vector_type(8)))  bf16  v8bf;
typedef __attribute__((ext_vector_type(16))) bf16  v16bf;
typedef __attribute__((ext_vector_type(8)))  float v8f;

#define B_   64
#define T_   512
#define D_   512
#define U_   1024
#define G4_  4096          // 4*U
#define NWG  64            // each WG owns 16 h-columns
#define WGSZ 128           // 4 waves (wave32) = 4 batch M-tiles

// LDS (elements of bf16): Wx frags [4 gates][16 ktiles], Wh frags [4][32]
// fragment = 32 lanes * 16 bf16 = 512 elems = 1024 bytes
#define LDS_WH_ELEM  (4 * 16 * 512)            // 32768 elems (= 64 KB) Wx region size
#define LDS_BYTES    ((4 * 16 * 512 + 4 * 32 * 512) * 2)   // 196608 bytes

// ---- hardware tanh (gfx1250 v_tanh_f32) with compile-safe fallback ----
#if defined(__has_builtin)
#  if __has_builtin(__builtin_amdgcn_tanhf)
#    define FAST_TANH(x) __builtin_amdgcn_tanhf(x)
#  elif __has_builtin(__builtin_amdgcn_tanh_f32)
#    define FAST_TANH(x) __builtin_amdgcn_tanh_f32(x)
#  else
#    define FAST_TANH(x) tanhf(x)
#  endif
#else
#  define FAST_TANH(x) tanhf(x)
#endif

__device__ __forceinline__ float sigmoidf_(float x) {
  // sigmoid(x) = 0.5*tanh(0.5x) + 0.5  -> 1 VALU fma + 1 TRANS + 1 fma
  return __builtin_fmaf(0.5f, FAST_TANH(0.5f * x), 0.5f);
}

// ---------------------------------------------------------------------------
// Init: zero barrier counter + h double-buffer, convert x -> bf16.
// ---------------------------------------------------------------------------
__global__ void lstm_init_kernel(const float* __restrict__ x,
                                 bf16* __restrict__ xb,
                                 bf16* __restrict__ hbuf,
                                 unsigned* __restrict__ cnt,
                                 int n) {
  const int stride = gridDim.x * blockDim.x;
  const int idx = blockIdx.x * blockDim.x + threadIdx.x;
  if (idx == 0) *cnt = 0u;
  for (int i = idx; i < 2 * B_ * U_; i += stride) hbuf[i] = (bf16)0.0f;
  for (int i = idx; i < n; i += stride) xb[i] = (bf16)x[i];
}

// ---------------------------------------------------------------------------
// Persistent LSTM kernel. grid = NWG blocks of WGSZ threads.
// ---------------------------------------------------------------------------
__global__ void __launch_bounds__(WGSZ, 1)
lstm_persistent(const bf16*  __restrict__ xb,    // [B, T, D] bf16
                const float* __restrict__ Wx,    // [D, 4U] f32
                const float* __restrict__ Wh,    // [U, 4U] f32
                const float* __restrict__ bias,  // [4U] f32
                float*       __restrict__ out,   // [B, T, U] f32
                bf16*        __restrict__ hbuf,  // [2, B, U] bf16 (double buffer)
                unsigned*    __restrict__ bar)   // grid barrier counter
{
  extern __shared__ char smem_raw[];
  bf16* smem = (bf16*)smem_raw;

  const int tid   = threadIdx.x;
  const int mt    = tid >> 5;          // wave id = batch M-tile (0..3)
  const int lane  = tid & 31;
  const int col16 = lane & 15;
  const int half  = (lane >> 4) & 1;
  const int ng    = blockIdx.x;        // 0..63 : owns h columns [ng*16, ng*16+16)

  // ---- one-time: gather+convert this WG's weight slice into LDS in WMMA
  //      B-fragment form, using this kernel's contiguous k order (lane L: col N=L%16; K = (L/16)*16 + j, j=0..15)
  for (int idx = tid; idx < 192 * 32; idx += WGSZ) {
    const int f     = idx >> 5;        // fragment id
    const int sl    = idx & 31;        // lane slot in fragment
    const int shalf = (sl >> 4) & 1;
    const int scol  = sl & 15;
    const float* W;
    int gate, kt, ldsoff;
    if (f < 64) {                                       // Wx region
      gate = f >> 4; kt = f & 15; W = Wx; ldsoff = f * 512;
    } else {                                            // Wh region
      const int f2 = f - 64;
      gate = f2 >> 5; kt = f2 & 31; W = Wh; ldsoff = LDS_WH_ELEM + f2 * 512;
    }
    const int ncol = gate * U_ + ng * 16 + scol;
    bf16* dst = smem + ldsoff + sl * 16;
#pragma unroll
    for (int j = 0; j < 16; ++j) {
      const int k = kt * 32 + shalf * 16 + j;
      dst[j] = (bf16)W[(size_t)k * G4_ + ncol];
    }
  }
  __syncthreads();

  // bias is per gate-column -> per-lane constant across all 8 acc rows
  float bias_g[4];
#pragma unroll
  for (int g = 0; g < 4; ++g) bias_g[g] = bias[g * U_ + ng * 16 + col16];

  const int arow  = mt * 16 + col16;   // A-fragment: batch row this lane holds
  const int ucol  = ng * 16 + col16;   // C/D: h column this lane holds
  const int brow0 = mt * 16 + half * 8;// C/D: first batch row of this lane

  v8f cstate;                           // cell state tile, lives in VGPRs
#pragma unroll
  for (int v = 0; v < 8; ++v) cstate[v] = 0.0f;

  const bf16* xrow = xb + (size_t)arow * (T_ * D_);

  for (int t = 0; t < T_; ++t) {
    const int p = t & 1;
    const bf16* hread  = hbuf + (size_t)p       * (B_ * U_);
    bf16*       hwrite = hbuf + (size_t)(p ^ 1) * (B_ * U_);

    v8f acc[4];
#pragma unroll
    for (int g = 0; g < 4; ++g)
#pragma unroll
      for (int v = 0; v < 8; ++v) acc[g][v] = bias_g[g];

    // ---- gates += x_t @ Wx  (K = 512) ----
    const bf16* xa = xrow + (size_t)t * D_;
#pragma unroll 2
    for (int kt = 0; kt < 16; ++kt) {
      const bf16* ap = xa + kt * 32 + half * 16;
      const v8bf alo = *(const v8bf*)(ap);
      const v8bf ahi = *(const v8bf*)(ap + 8);
      const v16bf a = __builtin_shufflevector(alo, ahi,
          0, 1, 2, 3, 4, 5, 6, 7, 8, 9, 10, 11, 12, 13, 14, 15);
#pragma unroll
      for (int g = 0; g < 4; ++g) {
        const v16bf bf = *(const v16bf*)(smem + (g * 16 + kt) * 512 + lane * 16);
        acc[g] = __builtin_amdgcn_wmma_f32_16x16x32_bf16(
            false, a, false, bf, (short)0, acc[g], false, false);
      }
    }

    // prefetch next timestep's x row into cache (global_prefetch_b8)
    if (t + 1 < T_) __builtin_prefetch(xa + D_, 0, 1);

    // ---- gates += h @ Wh  (K = 1024) ----
    const bf16* ha = hread + (size_t)arow * U_;
#pragma unroll 2
    for (int kt = 0; kt < 32; ++kt) {
      const bf16* ap = ha + kt * 32 + half * 16;
      const v8bf alo = *(const v8bf*)(ap);
      const v8bf ahi = *(const v8bf*)(ap + 8);
      const v16bf a = __builtin_shufflevector(alo, ahi,
          0, 1, 2, 3, 4, 5, 6, 7, 8, 9, 10, 11, 12, 13, 14, 15);
#pragma unroll
      for (int g = 0; g < 4; ++g) {
        const v16bf bf = *(const v16bf*)(smem + LDS_WH_ELEM + (g * 32 + kt) * 512 + lane * 16);
        acc[g] = __builtin_amdgcn_wmma_f32_16x16x32_bf16(
            false, a, false, bf, (short)0, acc[g], false, false);
      }
    }

    // ---- activations + state update (hardware v_tanh_f32) ----
    float hval[8];
#pragma unroll
    for (int v = 0; v < 8; ++v) {
      const float iv = sigmoidf_(acc[0][v]);
      const float fv = sigmoidf_(acc[1][v]);
      const float gv = FAST_TANH(acc[2][v]);
      const float ov = sigmoidf_(acc[3][v]);
      const float cv = fv * cstate[v] + iv * gv;
      cstate[v] = cv;
      hval[v] = ov * FAST_TANH(cv);
    }

    // ---- store h (cross-WG dependency) first, then the output slab ----
#pragma unroll
    for (int v = 0; v < 8; ++v)
      hwrite[(brow0 + v) * U_ + ucol] = (bf16)hval[v];
#pragma unroll
    for (int v = 0; v < 8; ++v)
      out[((size_t)(brow0 + v) * T_ + t) * U_ + ucol] = hval[v];

    // ---- grid-wide barrier: all h writes for step t visible before t+1 ----
    __syncthreads();
    if (tid == 0) {
      __threadfence();                              // release h stores
      atomicAdd(bar, 1u);
      const unsigned target = (unsigned)(t + 1) * NWG;
      while (__hip_atomic_load(bar, __ATOMIC_ACQUIRE, __HIP_MEMORY_SCOPE_AGENT) < target) {
        __builtin_amdgcn_s_sleep(2);
      }
    }
    __syncthreads();
    __threadfence();                                // acquire before reading h
  }
}

// ---------------------------------------------------------------------------
// Host-side launcher. Inputs: x[B,T,D], Wx[D,4U], Wh[U,4U], b[4U] (all f32).
// Output: out[B,T,U] f32.  ws: counter | h double buffer | x as bf16 (~34 MB).
// ---------------------------------------------------------------------------
extern "C" void kernel_launch(void* const* d_in, const int* in_sizes, int n_in,
                              void* d_out, int out_size, void* d_ws, size_t ws_size,
                              hipStream_t stream) {
  (void)in_sizes; (void)n_in; (void)out_size; (void)ws_size;

  const float* x  = (const float*)d_in[0];
  const float* Wx = (const float*)d_in[1];
  const float* Wh = (const float*)d_in[2];
  const float* b  = (const float*)d_in[3];
  float* out = (float*)d_out;

  char* ws = (char*)d_ws;
  unsigned* cnt = (unsigned*)ws;                              // @0
  bf16* hbuf = (bf16*)(ws + 4096);                            // 2*64*1024*2 = 256 KB
  bf16* xb   = (bf16*)(ws + 4096 + 2 * B_ * U_ * 2);          // 64*512*512*2 = 32 MB

  const int n_x = B_ * T_ * D_;
  lstm_init_kernel<<<4096, 256, 0, stream>>>(x, xb, hbuf, cnt, n_x);
  lstm_persistent<<<NWG, WGSZ, LDS_BYTES, stream>>>(xb, Wx, Wh, b, out, hbuf, cnt);
}
